// GPT_60799557042795
// MI455X (gfx1250) — matrix operand order repaired and checked by reading, not compiled, not run on hardware
//
#include <hip/hip_runtime.h>
#include <hip/hip_bf16.h>
#include <cstdint>
#include <cstddef>

// Problem dims (match reference)
#define B_  4
#define T_  2048
#define C_  1024
#define NH_ 16
#define HS_ 64
#define C3_ (3 * C_)

typedef __attribute__((ext_vector_type(16))) __bf16         v16bf;
typedef __attribute__((ext_vector_type(8)))  float          v8f;
typedef __attribute__((ext_vector_type(8)))  unsigned short u16x8;
typedef __attribute__((ext_vector_type(4)))  unsigned int   u32x4;
typedef __attribute__((ext_vector_type(8)))  int            i32x8;
typedef __attribute__((ext_vector_type(4)))  int            i32x4;

// ---------- helpers ----------
static __device__ __forceinline__ unsigned short f2bf_raw(float f) {
  unsigned u = __builtin_bit_cast(unsigned, f);
  unsigned r = u + 0x7FFFu + ((u >> 16) & 1u);   // round-to-nearest-even
  return (unsigned short)(r >> 16);
}
static __device__ __forceinline__ __bf16 bfc(unsigned short u) {
  return __builtin_bit_cast(__bf16, u);
}
static __device__ __forceinline__ v8f wmma_bf16(v16bf a, v16bf b, v8f c) {
  return __builtin_amdgcn_wmma_f32_16x16x32_bf16(false, a, false, b, (short)0, c,
                                                 false, false);
}
// low 32 bits of a generic pointer into LDS == LDS byte offset (aperture rule)
static __device__ __forceinline__ unsigned lds_off(const void* p) {
  return (unsigned)(unsigned long long)p;
}

// ---- Tensor Data Mover: 2-D tile Global->LDS (D# per CDNA5 ISA ch.8) ----
// dims/strides in bf16 elements (data_size = 2 bytes).
// 6-arg builtin: (uint32x4 g0, int32x8 g1, int32x4 g2, int32x4 g3, int32x8 g4, i32 cpol)
static __device__ __forceinline__ void tdm_load_2d(unsigned lds_addr,
                                                   const void* gaddr,
                                                   unsigned tile_w, unsigned tile_h,
                                                   unsigned long long row_stride,
                                                   unsigned tensor_w,
                                                   unsigned tensor_h) {
  unsigned long long ga = (unsigned long long)gaddr;
  u32x4 g0;
  g0[0] = 1u;                                        // count=1, user-mode load
  g0[1] = lds_addr;                                  // lds_addr[31:0]
  g0[2] = (unsigned)(ga & 0xFFFFFFFFu);              // global_addr[31:0]
  g0[3] = (unsigned)((ga >> 32) & 0x01FFFFFFu) | (2u << 30);  // ga[56:32] | type=2
  i32x8 g1;
  g1[0] = (int)(1u << 16);                                   // data_size=1 (2B)
  g1[1] = (int)((tensor_w & 0xFFFFu) << 16);                 // tensor_dim0[15:0]
  g1[2] = (int)((tensor_w >> 16) | ((tensor_h & 0xFFFFu) << 16));
  g1[3] = (int)((tensor_h >> 16) | ((tile_w & 0xFFFFu) << 16));
  g1[4] = (int)(tile_h & 0xFFFFu);                           // tile_dim1, tile_dim2=0
  g1[5] = (int)(unsigned)(row_stride & 0xFFFFFFFFull);       // dim0_stride[31:0]
  g1[6] = (int)(unsigned)((row_stride >> 32) & 0xFFFFull);   // dim0_stride[47:32]
  g1[7] = 0;
  i32x4 gz4 = {0, 0, 0, 0};
  i32x8 gz8 = {0, 0, 0, 0, 0, 0, 0, 0};
  __builtin_amdgcn_tensor_load_to_lds(g0, g1, gz4, gz4, gz8, 0);
}

// A fragment: 16x32 bf16, row-major source (row stride ld elements).
// Lane L(0-15): row M=L, K = half*16 + e (16 contiguous elements), half=lane>>4.
static __device__ __forceinline__ v16bf load_A16x32(const unsigned short* p, int ld,
                                                    int lane) {
  const int m = lane & 15, half = lane >> 4;
  const unsigned short* r = p + (size_t)m * ld + half * 16;
  u16x8 lo = *(const u16x8*)(r);
  u16x8 hi = *(const u16x8*)(r + 8);
  v16bf a;
#pragma unroll
  for (int i = 0; i < 8; ++i) { a[i] = bfc(lo[i]); a[8 + i] = bfc(hi[i]); }
  return a;
}

// B fragment: 32x16 from "n-major" source (row n = output col, K contiguous).
// Lane L: n = L&15, K = half*16 + e (16 contiguous elements).
static __device__ __forceinline__ v16bf load_B_kc(const unsigned short* p, int ld,
                                                  int lane) {
  const int n = lane & 15, half = lane >> 4;
  const unsigned short* r = p + (size_t)n * ld + half * 16;
  u16x8 lo = *(const u16x8*)(r);
  u16x8 hi = *(const u16x8*)(r + 8);
  v16bf b;
#pragma unroll
  for (int i = 0; i < 8; ++i) { b[i] = bfc(lo[i]); b[8 + i] = bfc(hi[i]); }
  return b;
}

// B fragment from k-major memory (rows = K dim, cols = N): per-lane K gather.
static __device__ __forceinline__ v16bf load_B_strided(const unsigned short* p, int ld,
                                                       int lane) {
  const int n = lane & 15, half = lane >> 4;
  const unsigned short* c = p + n + (size_t)(half * 16) * ld;
  v16bf b;
#pragma unroll
  for (int e = 0; e < 16; ++e) b[e] = bfc(c[(size_t)e * ld]);
  return b;
}

// ---------- pack / convert kernels ----------
__global__ void cvt_f32_bf16(const float* __restrict__ src,
                             unsigned short* __restrict__ dst, int n) {
  for (int i = blockIdx.x * blockDim.x + threadIdx.x; i < n;
       i += gridDim.x * blockDim.x)
    dst[i] = f2bf_raw(src[i]);
}

// src: R x Ccols row-major fp32 -> dst: Ccols x R ("n-major") bf16
__global__ void transpose_cvt(const float* __restrict__ src,
                              unsigned short* __restrict__ dst, int R, int Ccols) {
  int n = R * Ccols;
  for (int i = blockIdx.x * blockDim.x + threadIdx.x; i < n;
       i += gridDim.x * blockDim.x) {
    int nn = i / R, kk = i - nn * R;
    dst[i] = f2bf_raw(src[(size_t)kk * Ccols + nn]);
  }
}

// ---------- GEMM core: block stages A tile (16 x C) in LDS via TDM;
//   each wave computes 16x64; K-loop unrolled 2x with ping-pong B buffers
//   (no register rotation copies) ----------
template <int NCOLS>
static __device__ __forceinline__ void gemm_body(const unsigned short* __restrict__ Ag,
                                                 const unsigned short* __restrict__ Wn,
                                                 unsigned short* As,  // LDS, 16 x C_
                                                 int row0, int nt, int lane, int wslot,
                                                 v8f acc[4]) {
  if (wslot == 0) {
    tdm_load_2d(lds_off(As), Ag + (size_t)row0 * C_, C_, 16, C_, C_, 16);
    __builtin_amdgcn_s_wait_tensorcnt(0);
  }
  __syncthreads();

  const int n0 = nt * 64;
  const unsigned short* W0 = Wn + (size_t)n0 * C_;
  v16bf b0[4], b1[4];
#pragma unroll
  for (int j = 0; j < 4; ++j) b0[j] = load_B_kc(W0 + (size_t)(j * 16) * C_, C_, lane);

  for (int k0 = 0; k0 < C_; k0 += 64) {
    const int kA = (k0 + 32) & (C_ - 1);
    const int kB = (k0 + 64) & (C_ - 1);   // wrap keeps loop branch-free
    __builtin_prefetch(W0 + kB + (lane & 15) * 64, 0, 1);

    // phase 0: consume b0 (K = k0), fetch b1 (K = k0+32)
    v16bf a0 = load_A16x32(As + k0, C_, lane);
#pragma unroll
    for (int j = 0; j < 4; ++j)
      b1[j] = load_B_kc(W0 + (size_t)(j * 16) * C_ + kA, C_, lane);
#pragma unroll
    for (int j = 0; j < 4; ++j) acc[j] = wmma_bf16(a0, b0[j], acc[j]);

    // phase 1: consume b1 (K = k0+32), fetch b0 (K = k0+64)
    v16bf a1 = load_A16x32(As + kA, C_, lane);
#pragma unroll
    for (int j = 0; j < 4; ++j)
      b0[j] = load_B_kc(W0 + (size_t)(j * 16) * C_ + kB, C_, lane);
#pragma unroll
    for (int j = 0; j < 4; ++j) acc[j] = wmma_bf16(a1, b1[j], acc[j]);
  }
}

// QKV GEMM: (B*T x C) @ (C x 3C) -> bf16.  Block = (mt, group of 8 n-tiles).
__global__ __launch_bounds__(256) void qkv_gemm(const unsigned short* __restrict__ Xb,
                                                const unsigned short* __restrict__ WAb,
                                                unsigned short* __restrict__ QKV) {
  __shared__ __align__(16) unsigned short As[16 * C_];   // 32 KB
  const int lane = threadIdx.x & 31;
  const int wslot = threadIdx.x >> 5;
  const int mt = blockIdx.x / 6;          // 512 row tiles
  const int ng = blockIdx.x % 6;          // 6 groups of 8 n64-tiles (48 total)
  const int nt = ng * 8 + wslot;
  const int row0 = mt * 16;

  v8f acc[4] = {v8f{}, v8f{}, v8f{}, v8f{}};
  gemm_body<C3_>(Xb, WAb, As, row0, nt, lane, wslot, acc);

  const int half = lane >> 4, nn = lane & 15;
  const int n0 = nt * 64;
#pragma unroll
  for (int j = 0; j < 4; ++j)
#pragma unroll
    for (int r = 0; r < 8; ++r) {
      int m = r + half * 8;
      QKV[(size_t)(row0 + m) * C3_ + (n0 + j * 16 + nn)] = f2bf_raw(acc[j][r]);
    }
}

// Projection GEMM: (B*T x C) @ (C x C) -> fp32 out.
__global__ __launch_bounds__(256) void proj_gemm(const unsigned short* __restrict__ Yb,
                                                 const unsigned short* __restrict__ WPb,
                                                 float* __restrict__ out) {
  __shared__ __align__(16) unsigned short As[16 * C_];   // 32 KB
  const int lane = threadIdx.x & 31;
  const int wslot = threadIdx.x >> 5;
  const int mt = blockIdx.x / 2;          // 512 row tiles
  const int ng = blockIdx.x % 2;          // 2 groups of 8 n64-tiles (16 total)
  const int nt = ng * 8 + wslot;
  const int row0 = mt * 16;

  v8f acc[4] = {v8f{}, v8f{}, v8f{}, v8f{}};
  gemm_body<C_>(Yb, WPb, As, row0, nt, lane, wslot, acc);

  const int half = lane >> 4, nn = lane & 15;
  const int n0 = nt * 64;
#pragma unroll
  for (int j = 0; j < 4; ++j)
#pragma unroll
    for (int r = 0; r < 8; ++r) {
      int m = r + half * 8;
      out[(size_t)(row0 + m) * C_ + (n0 + j * 16 + nn)] = acc[j][r];
    }
}

// ---------- flash attention ----------
// Block: 8 waves share (b,h), handle 8 consecutive 16-query tiles.
// K/V 32x64 tiles are staged cooperatively into LDS via TDM each key-block.
__global__ __launch_bounds__(256) void attn(const unsigned short* __restrict__ QKV,
                                            unsigned short* __restrict__ Yb) {
  __shared__ __align__(16) unsigned short Kt[32 * HS_];      // 4 KB
  __shared__ __align__(16) unsigned short Vt[32 * HS_];      // 4 KB
  __shared__ __align__(16) unsigned short pbuf[8 * 16 * 32]; // 8 KB
  const int lane = threadIdx.x & 31;
  const int wslot = threadIdx.x >> 5;
  const int bh = blockIdx.x >> 4;           // 64 (b,h) pairs
  const int qtg = blockIdx.x & 15;          // 16 query-tile groups per (b,h)
  const int b = bh >> 4;
  const int h = bh & 15;
  const int qt = qtg * 8 + wslot;
  const int q0 = qt * 16;
  unsigned short* P = pbuf + wslot * (16 * 32);

  const unsigned short* Qp = QKV + (size_t)(b * T_ + q0) * C3_ + h * HS_;
  const v16bf aq0 = load_A16x32(Qp, C3_, lane);
  const v16bf aq1 = load_A16x32(Qp + 32, C3_, lane);

  v8f o[4] = {v8f{}, v8f{}, v8f{}, v8f{}};
  float mrow[8], lrow[8];
#pragma unroll
  for (int r = 0; r < 8; ++r) { mrow[r] = -1e30f; lrow[r] = 0.0f; }

  const float scale = 0.125f;  // 1/sqrt(64)
  const int half = lane >> 4, nn = lane & 15;
  const int my_nblk = (q0 + 16 + 31) / 32;
  const int nblk_max = ((qtg * 8 + 7) * 16 + 16 + 31) / 32;  // block-uniform

  for (int kb = 0; kb < nblk_max; ++kb) {
    const int t0 = kb * 32;
    __syncthreads();  // previous iteration's K/V reads complete
    if (wslot == 0) {
      const unsigned short* Kg = QKV + (size_t)(b * T_ + t0) * C3_ + C_ + h * HS_;
      const unsigned short* Vg = QKV + (size_t)(b * T_ + t0) * C3_ + 2 * C_ + h * HS_;
      tdm_load_2d(lds_off(Kt), Kg, HS_, 32, C3_, HS_, 32);
      tdm_load_2d(lds_off(Vt), Vg, HS_, 32, C3_, HS_, 32);
      __builtin_amdgcn_s_wait_tensorcnt(0);
    }
    __syncthreads();  // K/V tile visible to all waves

    if (kb < my_nblk) {
      // S = Q * K^T  (two 16-key column tiles) -- K^T fragments from LDS
      v8f s0{}, s1{};
      s0 = wmma_bf16(aq0, load_B_kc(Kt, HS_, lane), s0);
      s0 = wmma_bf16(aq1, load_B_kc(Kt + 32, HS_, lane), s0);
      s1 = wmma_bf16(aq0, load_B_kc(Kt + 16 * HS_, HS_, lane), s1);
      s1 = wmma_bf16(aq1, load_B_kc(Kt + 16 * HS_ + 32, HS_, lane), s1);

      // online softmax; C-frag element (r,lane): M = r + half*8, N = nn
#pragma unroll
      for (int r = 0; r < 8; ++r) {
        const int m = r + half * 8;
        const int q = q0 + m;
        float v0 = s0[r] * scale, v1 = s1[r] * scale;
        if (t0 + nn > q) v0 = -1e30f;
        if (t0 + 16 + nn > q) v1 = -1e30f;
        float bm = fmaxf(v0, v1);
#pragma unroll
        for (int sh = 1; sh < 16; sh <<= 1) bm = fmaxf(bm, __shfl_xor(bm, sh, 32));
        const float mnew = fmaxf(mrow[r], bm);
        const float alpha = __expf(mrow[r] - mnew);
        const float p0 = __expf(v0 - mnew);
        const float p1 = __expf(v1 - mnew);
        float rs = p0 + p1;
#pragma unroll
        for (int sh = 1; sh < 16; sh <<= 1) rs += __shfl_xor(rs, sh, 32);
        lrow[r] = lrow[r] * alpha + rs;
        mrow[r] = mnew;
#pragma unroll
        for (int d = 0; d < 4; ++d) o[d][r] = o[d][r] * alpha;
        P[m * 32 + nn] = f2bf_raw(p0);
        P[m * 32 + 16 + nn] = f2bf_raw(p1);
      }
      // per-wave cross-lane LDS handoff; DS ops are in-order per wave
      __asm__ volatile("s_wait_dscnt 0" ::: "memory");
      const v16bf pa = load_A16x32(P, 32, lane);

      // O += P * V  (V tile in LDS, k-major -> strided gather)
#pragma unroll
      for (int d = 0; d < 4; ++d) {
        v16bf bv = load_B_strided(Vt + d * 16, HS_, lane);
        o[d] = wmma_bf16(pa, bv, o[d]);
      }
    }
  }

  // normalize and write y (B*T x C) bf16
#pragma unroll
  for (int r = 0; r < 8; ++r) {
    const int m = r + half * 8;
    const float inv = 1.0f / lrow[r];
#pragma unroll
    for (int d = 0; d < 4; ++d)
      Yb[(size_t)(b * T_ + q0 + m) * C_ + h * HS_ + d * 16 + nn] =
          f2bf_raw(o[d][r] * inv);
  }
}

// ---------- host-side launch ----------
extern "C" void kernel_launch(void* const* d_in, const int* in_sizes, int n_in,
                              void* d_out, int out_size, void* d_ws, size_t ws_size,
                              hipStream_t stream) {
  (void)in_sizes; (void)n_in; (void)out_size; (void)ws_size;
  const float* x  = (const float*)d_in[0];   // (B,T,C)
  const float* Wa = (const float*)d_in[1];   // (C,3C)
  const float* Wp = (const float*)d_in[2];   // (C,C)
  float* out = (float*)d_out;                // (B,T,C) fp32

  char* ws = (char*)d_ws;
  unsigned short* Xb  = (unsigned short*)(ws + 0);          // 16 MiB  x  bf16
  unsigned short* WAb = (unsigned short*)(ws + 16777216);   //  6 MiB  W_attn^T bf16
  unsigned short* WPb = (unsigned short*)(ws + 23068672);   //  2 MiB  W_proj^T bf16
  unsigned short* QKV = (unsigned short*)(ws + 25165824);   // 48 MiB  qkv bf16
  unsigned short* Yb  = (unsigned short*)(ws + 75497472);   // 16 MiB  attn out bf16

  cvt_f32_bf16<<<4096, 256, 0, stream>>>(x, Xb, B_ * T_ * C_);
  transpose_cvt<<<2048, 256, 0, stream>>>(Wa, WAb, C_, C3_);
  transpose_cvt<<<1024, 256, 0, stream>>>(Wp, WPb, C_, C_);

  qkv_gemm<<<3072, 256, 0, stream>>>(Xb, WAb, QKV);   // 512 mt x 6 ngroups
  attn<<<1024, 256, 0, stream>>>(QKV, Yb);            // 64 (b,h) x 16 qt-groups
  proj_gemm<<<1024, 256, 0, stream>>>(Yb, WPb, out);  // 512 mt x 2 ngroups
}
